// HybridKernel_65481071409271
// MI455X (gfx1250) — matrix operand order repaired and checked by reading, not compiled, not run on hardware
//
#include <hip/hip_runtime.h>

typedef __bf16        v16bf __attribute__((ext_vector_type(16)));
typedef float         v8f   __attribute__((ext_vector_type(8)));
typedef float         f32x4 __attribute__((ext_vector_type(4)));
typedef unsigned int  u32x2 __attribute__((ext_vector_type(2)));
typedef unsigned int  u32x4 __attribute__((ext_vector_type(4)));
typedef unsigned int  u32x8 __attribute__((ext_vector_type(8)));

#define TILE    128   // output tile per workgroup: 128x128
#define KDIM    128   // full K (fits in LDS, single load phase)
#define LSTRIDE 136   // padded LDS row stride (bf16 elems): 272B -> 4-bank rotation/row

__device__ __forceinline__ unsigned f2bf(float f) {
  unsigned u = __float_as_uint(f);
  return (u + 0x7FFFu + ((u >> 16) & 1u)) >> 16;   // RNE f32 -> bf16
}
__device__ __forceinline__ float bf2f(unsigned h) {
  return __uint_as_float(h << 16);
}

// A operand (16x32 bf16): lane grp g -> VGPR0..3 = K[kb+8g .. +7], VGPR4..7 = K[kb+16+8g .. +7]
__device__ __forceinline__ v16bf load_a_op(const unsigned short* s, int row, int kb, int grp) {
  const unsigned short* p = s + row * LSTRIDE + kb + grp * 8;
  u32x8 w;
  w.lo = *(const u32x4*)(p);        // K .. K+7
  w.hi = *(const u32x4*)(p + 16);   // K+16 .. K+23
  return __builtin_bit_cast(v16bf, w);
}
// B operand (32x16 bf16): lane grp g -> VGPR0..3 = K[kb+8g .. +7], VGPR4..7 = K[kb+16+8g .. +7]
__device__ __forceinline__ v16bf load_b_op(const unsigned short* s, int col, int kb, int grp) {
  const unsigned short* p = s + col * LSTRIDE + kb + grp * 8;
  u32x8 w;
  w.lo = *(const u32x4*)(p);        // K .. K+7
  w.hi = *(const u32x4*)(p + 16);   // K+16 .. K+23
  return __builtin_bit_cast(v16bf, w);
}

__global__ __launch_bounds__(256)
void rbf_gram_wmma(const float* __restrict__ A, const float* __restrict__ B,
                   float* __restrict__ out, int n, int m) {
  __shared__ unsigned short sAhi[TILE * LSTRIDE];
  __shared__ unsigned short sAlo[TILE * LSTRIDE];
  __shared__ unsigned short sBhi[TILE * LSTRIDE];
  __shared__ unsigned short sBlo[TILE * LSTRIDE];
  __shared__ float sAn[TILE];
  __shared__ float sBn[TILE];

  const int t     = threadIdx.x;
  const int mBase = blockIdx.y * TILE;   // rows  <- a
  const int nBase = blockIdx.x * TILE;   // cols  <- b

  // ---- Phase 1: load f32 rows, split into bf16 hi/lo, compute row norms ----
  {
    const int  r   = t & 127;
    const bool isB = (t >= 128);
    const float* row = (isB ? B : A) + (size_t)((isB ? nBase : mBase) + r) * KDIM;
    unsigned short* hi = isB ? sBhi : sAhi;
    unsigned short* lo = isB ? sBlo : sAlo;
    float nrm = 0.0f;
#pragma unroll
    for (int k = 0; k < KDIM; k += 4) {
      f32x4 v = *(const f32x4*)(row + k);
      unsigned hp[4], lp[4];
#pragma unroll
      for (int i = 0; i < 4; ++i) {
        float x = v[i];
        nrm = fmaf(x, x, nrm);
        unsigned h = f2bf(x);
        hp[i] = h;
        lp[i] = f2bf(x - bf2f(h));
      }
      u32x2 ph, pl;
      ph.x = hp[0] | (hp[1] << 16);  ph.y = hp[2] | (hp[3] << 16);
      pl.x = lp[0] | (lp[1] << 16);  pl.y = lp[2] | (lp[3] << 16);
      *(u32x2*)(hi + r * LSTRIDE + k) = ph;
      *(u32x2*)(lo + r * LSTRIDE + k) = pl;
    }
    (isB ? sBn : sAn)[r] = nrm;
  }
  __syncthreads();

  // ---- Phase 2: split-bf16 WMMA GEMM (hi*hi + hi*lo + lo*hi) ----
  const int wave = t >> 5;
  const int lane = t & 31;
  const int wm   = wave & 3;    // 4 waves along M: 32 rows each
  const int wn   = wave >> 2;   // 2 waves along N: 64 cols each
  const int lrow = lane & 15;
  const int lgrp = lane >> 4;

  v8f acc[2][4];
#pragma unroll
  for (int i = 0; i < 2; ++i)
#pragma unroll
    for (int j = 0; j < 4; ++j)
      acc[i][j] = {};

#pragma unroll
  for (int kk = 0; kk < 4; ++kk) {
    const int kb = kk * 32;
    v16bf ahi[2], alo[2];
#pragma unroll
    for (int mt = 0; mt < 2; ++mt) {
      const int rrow = wm * 32 + mt * 16 + lrow;
      ahi[mt] = load_a_op(sAhi, rrow, kb, lgrp);
      alo[mt] = load_a_op(sAlo, rrow, kb, lgrp);
    }
#pragma unroll
    for (int nt = 0; nt < 4; ++nt) {
      const int ccol = wn * 64 + nt * 16 + lrow;
      v16bf bhi = load_b_op(sBhi, ccol, kb, lgrp);
      v16bf blo = load_b_op(sBlo, ccol, kb, lgrp);
#pragma unroll
      for (int mt = 0; mt < 2; ++mt) {
        acc[mt][nt] = __builtin_amdgcn_wmma_f32_16x16x32_bf16(
            false, ahi[mt], false, bhi, (short)0, acc[mt][nt], false, false);
        acc[mt][nt] = __builtin_amdgcn_wmma_f32_16x16x32_bf16(
            false, ahi[mt], false, blo, (short)0, acc[mt][nt], false, false);
        acc[mt][nt] = __builtin_amdgcn_wmma_f32_16x16x32_bf16(
            false, alo[mt], false, bhi, (short)0, acc[mt][nt], false, false);
      }
    }
  }

  // ---- Epilogue: d2 = |a|^2 + |b|^2 - 2 a.b ; out = exp(-gamma * d2), gamma=1 ----
  // Output is a 256MB write-once stream: use non-temporal stores so it doesn't
  // evict the L2-resident a/b working set (CDNA5 store TH=NT policy).
#pragma unroll
  for (int mt = 0; mt < 2; ++mt) {
#pragma unroll
    for (int nt = 0; nt < 4; ++nt) {
      const int   ln = wn * 64 + nt * 16 + lrow;
      const float bn = sBn[ln];
      const size_t gn = (size_t)(nBase + ln);
#pragma unroll
      for (int j = 0; j < 8; ++j) {
        const int   lm = wm * 32 + mt * 16 + j + 8 * lgrp;   // C/D layout: M = j + 8*grp
        const float d  = sAn[lm] + bn - 2.0f * acc[mt][nt][j];
        const size_t gm = (size_t)(mBase + lm);
        __builtin_nontemporal_store(__expf(-fmaxf(d, 0.0f)),
                                    out + gm * (size_t)m + gn);
      }
    }
  }
}

extern "C" void kernel_launch(void* const* d_in, const int* in_sizes, int n_in,
                              void* d_out, int out_size, void* d_ws, size_t ws_size,
                              hipStream_t stream) {
  const float* a = (const float*)d_in[0];
  const float* b = (const float*)d_in[1];
  float* out = (float*)d_out;
  const int n = in_sizes[0] / KDIM;   // 8192 rows of a
  const int m = in_sizes[1] / KDIM;   // 8192 rows of b
  dim3 grid(m / TILE, n / TILE);      // 64 x 64 workgroups
  rbf_gram_wmma<<<grid, dim3(256), 0, stream>>>(a, b, out, n, m);
  (void)n_in; (void)out_size; (void)d_ws; (void)ws_size;
}
